// SpGAT_44169443672088
// MI455X (gfx1250) — matrix operand order repaired and checked by reading, compiled, not run on hardware
//
#include <hip/hip_runtime.h>
#include <math.h>

typedef __attribute__((ext_vector_type(16))) _Float16 v16h;
typedef __attribute__((ext_vector_type(8)))  float    v8f;

#define LRELU_ALPHA 0.2f

// ---------------------------------------------------------------------------
// Prep kernels
// ---------------------------------------------------------------------------
__global__ void k_f32_to_f16(const float* __restrict__ src,
                             _Float16* __restrict__ dst, int n) {
  int t = blockIdx.x * blockDim.x + threadIdx.x;
  if (t < n) dst[t] = (_Float16)src[t];
}

// W1: [8][256][64] -> w1t[c*256 + k] = W1[hd][k][n], c = hd*64+n  (K-contiguous B)
__global__ void k_build_w1t(const float* __restrict__ W1,
                            _Float16* __restrict__ w1t) {
  int t = blockIdx.x * blockDim.x + threadIdx.x;
  if (t >= 512 * 256) return;
  int c = t >> 8, k = t & 255;
  int hd = c >> 6, n = c & 63;
  w1t[t] = (_Float16)W1[hd * (256 * 64) + k * 64 + n];
}

// W2: [512][40] -> w2t[c*512 + k] = (c<40) ? W2[k*40 + c] : 0   (pad cols to 64)
__global__ void k_build_w2t(const float* __restrict__ W2,
                            _Float16* __restrict__ w2t) {
  int t = blockIdx.x * blockDim.x + threadIdx.x;
  if (t >= 64 * 512) return;
  int c = t >> 9, k = t & 511;
  w2t[t] = (_Float16)((c < 40) ? W2[k * 40 + c] : 0.0f);
}

__global__ void k_build_bias2(const float* __restrict__ b2,
                              float* __restrict__ bias2) {
  int t = threadIdx.x;
  if (t < 64) bias2[t] = (t < 40) ? b2[t] : 0.0f;
}

// ---------------------------------------------------------------------------
// WMMA GEMM:  C[M][Nc] = A[M][K](f16) * BT[Nc][K](f16, pre-transposed) + bias
// One wave computes a 16x64 strip (1 M-tile x 4 N-tiles): the A fragment is
// loaded once per K-step and fed to 4 consecutive v_wmma_f32_16x16x32_f16,
// raising arithmetic intensity 1.6x vs one-tile-per-wave.
// Requires ntiles % 4 == 0 (GEMM1: 32, GEMM2: 4).
// ---------------------------------------------------------------------------
union frag16 { v16h h; unsigned int u[8]; };

__global__ void k_wmma_gemm(const _Float16* __restrict__ A,
                            const _Float16* __restrict__ BT,
                            const float* __restrict__ bias,
                            float* __restrict__ C,
                            int mtiles, int ngroups, int K, int Nc) {
  int wave = (blockIdx.x * blockDim.x + threadIdx.x) >> 5;
  int lane = threadIdx.x & 31;
  if (wave >= mtiles * ngroups) return;
  int mt = wave / ngroups, ng = wave % ngroups;
  int hf = lane >> 4;          // lane half (0: lanes 0-15, 1: lanes 16-31)
  int r  = lane & 15;

  const _Float16* arow = A + (size_t)(mt * 16 + r) * K;   // A row m0+r
  const _Float16* brow[4];
#pragma unroll
  for (int j = 0; j < 4; ++j)                             // B col (ng*64 + j*16 + r)
    brow[j] = BT + (size_t)(ng * 64 + j * 16 + r) * K;

  v8f acc[4] = {};
  for (int k0 = 0; k0 < K; k0 += 32) {
    frag16 a;
#pragma unroll
    for (int v = 0; v < 8; ++v) {
      int ka = k0 + ((v < 4) ? 0 : 16) + hf * 8 + 2 * (v & 3); // A 16x32 layout
      a.u[v] = *(const unsigned int*)(arow + ka);
    }
#pragma unroll
    for (int j = 0; j < 4; ++j) {
      frag16 b;
#pragma unroll
      for (int v = 0; v < 8; ++v) {
        int kb = k0 + ((v < 4) ? 0 : 16) + hf * 8 + 2 * (v & 3); // B 32x16 layout
        b.u[v] = *(const unsigned int*)(brow[j] + kb);
      }
      acc[j] = __builtin_amdgcn_wmma_f32_16x16x32_f16(
          false, a.h, false, b.h, (short)0, acc[j], false, false);
    }
  }

  // store: row = m0 + hf*8 + rr, col = ng*64 + j*16 + r
#pragma unroll
  for (int j = 0; j < 4; ++j) {
    int col = ng * 64 + j * 16 + r;
    float bv = bias[col];
    size_t base = (size_t)(mt * 16 + hf * 8) * Nc + col;
#pragma unroll
    for (int rr = 0; rr < 8; ++rr)
      C[base + (size_t)rr * Nc] = acc[j][rr] + bv;
  }
}

// ---------------------------------------------------------------------------
// Layer-1 attention scores: ssrc[hd*N+i] = h1[i,hd,:]·a1[hd,:64], sdst with a1[hd,64:]
// ---------------------------------------------------------------------------
__global__ void k_scores1(const float* __restrict__ h1,
                          const float* __restrict__ a1,
                          float* __restrict__ ssrc, float* __restrict__ sdst,
                          int N) {
  int t = blockIdx.x * blockDim.x + threadIdx.x;
  if (t >= N * 8) return;
  int i = t >> 3, hd = t & 7;
  const float* hp = h1 + (size_t)i * 512 + hd * 64;
  const float* av = a1 + hd * 128;
  float s0 = 0.f, s1 = 0.f;
#pragma unroll 8
  for (int f = 0; f < 64; ++f) {
    float v = hp[f];
    s0 += v * av[f];
    s1 += v * av[64 + f];
  }
  ssrc[hd * N + i] = s0;
  sdst[hd * N + i] = s1;
}

// Per-edge, per-head attention weight + rowsum
__global__ void k_edge_att1(const int* __restrict__ esrc, const int* __restrict__ edst,
                            const float* __restrict__ ssrc, const float* __restrict__ sdst,
                            float* __restrict__ ew, float* __restrict__ rowsum,
                            int E, int N) {
  long long t = (long long)blockIdx.x * blockDim.x + threadIdx.x;
  if (t >= (long long)E * 8) return;
  int e = (int)(t >> 3), hd = (int)(t & 7);
  int s = esrc[e], d = edst[e];
  float z  = ssrc[hd * N + s] + sdst[hd * N + d];
  float lz = (z > 0.f) ? z : LRELU_ALPHA * z;
  float w  = expf(lz);
  ew[(size_t)e * 8 + hd] = w;
  atomicAdd(&rowsum[hd * N + s], w);
}

// Edge feature aggregation: hprime[src] += w * h1[dst]   (2 edges / 256-thr block)
__global__ void k_edge_agg1(const int* __restrict__ esrc, const int* __restrict__ edst,
                            const float* __restrict__ ew, const float* __restrict__ h1,
                            float* __restrict__ hprime, int E) {
  int t = threadIdx.x;
  int e = blockIdx.x * 2 + (t >> 7);
  if (e >= E) return;
  int c = (t & 127) * 4;                 // 128 threads cover 512 floats
  int s = esrc[e], d = edst[e];
  float w = ew[(size_t)e * 8 + (c >> 6)];
  const float4 hv = *reinterpret_cast<const float4*>(h1 + (size_t)d * 512 + c);
  float* out = hprime + (size_t)s * 512 + c;
  atomicAdd(out + 0, w * hv.x);
  atomicAdd(out + 1, w * hv.y);
  atomicAdd(out + 2, w * hv.z);
  atomicAdd(out + 3, w * hv.w);
}

// Normalize + ELU + convert to f16 (A operand of GEMM2)
__global__ void k_finalize1(const float* __restrict__ hprime,
                            const float* __restrict__ rowsum,
                            _Float16* __restrict__ hcat16, int N) {
  int t = blockIdx.x * blockDim.x + threadIdx.x;
  if (t >= N * 512) return;
  int i = t >> 9, c = t & 511, hd = c >> 6;
  float v = hprime[t] / rowsum[hd * N + i];
  v = (v > 0.f) ? v : expm1f(v);          // ELU, alpha = 1
  hcat16[t] = (_Float16)v;
}

// Layer-2 scores (F = 40)
__global__ void k_scores2(const float* __restrict__ h2, const float* __restrict__ a2,
                          float* __restrict__ ssrc, float* __restrict__ sdst, int N) {
  int i = blockIdx.x * blockDim.x + threadIdx.x;
  if (i >= N) return;
  float s0 = 0.f, s1 = 0.f;
#pragma unroll 8
  for (int c = 0; c < 40; ++c) {
    float v = h2[(size_t)i * 64 + c];
    s0 += v * a2[c];
    s1 += v * a2[40 + c];
  }
  ssrc[i] = s0;
  sdst[i] = s1;
}

__global__ void k_edge_att2(const int* __restrict__ esrc, const int* __restrict__ edst,
                            const float* __restrict__ ssrc, const float* __restrict__ sdst,
                            float* __restrict__ ew2, float* __restrict__ rowsum2, int E) {
  int e = blockIdx.x * blockDim.x + threadIdx.x;
  if (e >= E) return;
  int s = esrc[e], d = edst[e];
  float z  = ssrc[s] + sdst[d];
  float lz = (z > 0.f) ? z : LRELU_ALPHA * z;
  float w  = expf(lz);
  ew2[e] = w;
  atomicAdd(&rowsum2[s], w);
}

// 64 threads per edge (cols 40..63 idle): outacc[src] += w * h2[dst]
__global__ void k_edge_agg2(const int* __restrict__ esrc, const int* __restrict__ edst,
                            const float* __restrict__ ew2, const float* __restrict__ h2,
                            float* __restrict__ outacc, int E) {
  int t = threadIdx.x;
  int e = blockIdx.x * 4 + (t >> 6);
  int c = t & 63;
  if (e >= E || c >= 40) return;
  int s = esrc[e], d = edst[e];
  atomicAdd(&outacc[(size_t)s * 64 + c], ew2[e] * h2[(size_t)d * 64 + c]);
}

// Normalize + log_softmax over 40 classes; one wave32 per node
__global__ void k_logsoftmax(const float* __restrict__ outacc,
                             const float* __restrict__ rowsum2,
                             float* __restrict__ out, int N) {
  int wid  = (blockIdx.x * blockDim.x + threadIdx.x) >> 5;
  int lane = threadIdx.x & 31;
  if (wid >= N) return;
  float inv = 1.0f / rowsum2[wid];
  float v0 = outacc[(size_t)wid * 64 + lane] * inv;                  // lane < 32 < 40
  bool  has1 = (lane + 32) < 40;                                     // lanes 0..7
  float v1 = has1 ? outacc[(size_t)wid * 64 + lane + 32] * inv : -INFINITY;
  float m = fmaxf(v0, v1);
  for (int o = 16; o > 0; o >>= 1) m = fmaxf(m, __shfl_xor(m, o, 32));
  float s = expf(v0 - m) + (has1 ? expf(v1 - m) : 0.0f);
  for (int o = 16; o > 0; o >>= 1) s += __shfl_xor(s, o, 32);
  float lse = m + logf(s);
  out[(size_t)wid * 40 + lane] = v0 - lse;
  if (has1) out[(size_t)wid * 40 + lane + 32] = v1 - lse;
}

// ---------------------------------------------------------------------------
// Host launcher
// ---------------------------------------------------------------------------
extern "C" void kernel_launch(void* const* d_in, const int* in_sizes, int n_in,
                              void* d_out, int out_size, void* d_ws, size_t ws_size,
                              hipStream_t stream) {
  const float* x   = (const float*)d_in[0];
  const int*   edg = (const int*)  d_in[1];
  const float* W1  = (const float*)d_in[2];
  const float* b1  = (const float*)d_in[3];   // [512] flat == per-column bias of GEMM1
  const float* a1  = (const float*)d_in[4];
  const float* W2  = (const float*)d_in[5];
  const float* b2  = (const float*)d_in[6];
  const float* a2  = (const float*)d_in[7];
  float* out = (float*)d_out;

  const int N = in_sizes[0] / 256;   // 50000
  const int E = in_sizes[1] / 2;     // 1650000
  const int* esrc = edg;
  const int* edst = edg + E;

  // --- workspace suballocation (256B aligned) ---
  char* p = (char*)d_ws;
  auto alloc = [&](size_t bytes) -> char* {
    char* r = p;
    p += (bytes + 255) & ~(size_t)255;
    return r;
  };
  _Float16* xh     = (_Float16*)alloc((size_t)N * 256 * 2);
  _Float16* w1t    = (_Float16*)alloc((size_t)512 * 256 * 2);
  _Float16* w2t    = (_Float16*)alloc((size_t)64 * 512 * 2);
  float*    bias2  = (float*)   alloc(64 * 4);
  float*    h1     = (float*)   alloc((size_t)N * 512 * 4);
  float*    ssrc1  = (float*)   alloc((size_t)8 * N * 4);
  float*    sdst1  = (float*)   alloc((size_t)8 * N * 4);
  float*    rowsum1= (float*)   alloc((size_t)8 * N * 4);
  float*    ew1    = (float*)   alloc((size_t)E * 8 * 4);
  float*    hprime = (float*)   alloc((size_t)N * 512 * 4);
  _Float16* hcat16 = (_Float16*)alloc((size_t)N * 512 * 2);
  float*    h2     = (float*)   alloc((size_t)N * 64 * 4);
  float*    ssrc2  = (float*)   alloc((size_t)N * 4);
  float*    sdst2  = (float*)   alloc((size_t)N * 4);
  float*    rowsum2= (float*)   alloc((size_t)N * 4);
  float*    ew2    = (float*)   alloc((size_t)E * 4);
  float*    outacc = (float*)   alloc((size_t)N * 64 * 4);
  (void)ws_size; (void)n_in; (void)out_size;

  const int B = 256;
  auto cdiv = [](long long a, long long b) { return (int)((a + b - 1) / b); };

  // zero atomic accumulators (capturable memset nodes)
  hipMemsetAsync(rowsum1, 0, (size_t)8 * N * 4, stream);
  hipMemsetAsync(hprime,  0, (size_t)N * 512 * 4, stream);
  hipMemsetAsync(rowsum2, 0, (size_t)N * 4, stream);
  hipMemsetAsync(outacc,  0, (size_t)N * 64 * 4, stream);

  // prep
  k_f32_to_f16<<<cdiv((long long)N * 256, B), B, 0, stream>>>(x, xh, N * 256);
  k_build_w1t <<<cdiv(512 * 256, B), B, 0, stream>>>(W1, w1t);
  k_build_w2t <<<cdiv(64 * 512, B), B, 0, stream>>>(W2, w2t);
  k_build_bias2<<<1, 64, 0, stream>>>(b2, bias2);

  // GEMM1: h1[N,512] = x @ W1 + b1   (mtiles=3125, ngroups=8, K=256)
  {
    int mt = N / 16, ng = 512 / 64, waves = mt * ng;
    k_wmma_gemm<<<cdiv((long long)waves * 32, B), B, 0, stream>>>(
        xh, w1t, b1, h1, mt, ng, 256, 512);
  }

  // layer-1 attention
  k_scores1<<<cdiv((long long)N * 8, B), B, 0, stream>>>(h1, a1, ssrc1, sdst1, N);
  k_edge_att1<<<cdiv((long long)E * 8, B), B, 0, stream>>>(
      esrc, edst, ssrc1, sdst1, ew1, rowsum1, E, N);
  k_edge_agg1<<<cdiv(E, 2), B, 0, stream>>>(esrc, edst, ew1, h1, hprime, E);
  k_finalize1<<<cdiv((long long)N * 512, B), B, 0, stream>>>(hprime, rowsum1, hcat16, N);

  // GEMM2: h2[N,64] = hcat @ W2 + b2 (cols 40..63 exactly zero; mtiles=3125, ngroups=1)
  {
    int mt = N / 16, ng = 64 / 64, waves = mt * ng;
    k_wmma_gemm<<<cdiv((long long)waves * 32, B), B, 0, stream>>>(
        hcat16, w2t, bias2, h2, mt, ng, 512, 64);
  }

  // layer-2 attention
  k_scores2<<<cdiv(N, B), B, 0, stream>>>(h2, a2, ssrc2, sdst2, N);
  k_edge_att2<<<cdiv(E, B), B, 0, stream>>>(esrc, edst, ssrc2, sdst2, ew2, rowsum2, E);
  k_edge_agg2<<<cdiv(E, 4), B, 0, stream>>>(esrc, edst, ew2, h2, outacc, E);

  // normalize + log_softmax -> d_out [N,40]
  k_logsoftmax<<<cdiv((long long)N * 32, B), B, 0, stream>>>(outacc, rowsum2, out, N);
}
